// MaskMamba_3_32882269618434
// MI455X (gfx1250) — matrix operand order repaired and checked by reading, not compiled, not run on hardware
//
#include <hip/hip_runtime.h>
#include <hip/hip_bf16.h>

// ---------------------------------------------------------------------------
// Point-Mamba forward for MI455X (gfx1250, wave32, WMMA).
// All GEMMs run as v_wmma_f32_16x16x32_bf16 (bf16 in, f32 accumulate).
// GEMM A-tiles are staged LDS-direct via GLOBAL_LOAD_ASYNC_TO_LDS_B128
// (ASYNCcnt double-buffering) when the toolchain exposes the builtin.
// ---------------------------------------------------------------------------

typedef __attribute__((ext_vector_type(16))) __bf16 v16bf;
typedef __attribute__((ext_vector_type(8)))  float  v8f;
typedef __attribute__((ext_vector_type(4)))  int    v4i;

union BFrag { v16bf v; uint4 q[2]; };

#define AS1 __attribute__((address_space(1)))
#define AS3 __attribute__((address_space(3)))

#if defined(__HIP_DEVICE_COMPILE__) &&                                        \
    __has_builtin(__builtin_amdgcn_global_load_async_to_lds_b128) &&          \
    __has_builtin(__builtin_amdgcn_s_wait_asynccnt)
#define GEMM_ASYNC 1
#else
#define GEMM_ASYNC 0
#endif

__device__ __forceinline__ unsigned short f2bf(float f) {
    unsigned int u = __float_as_uint(f);
    u += 0x7FFFu + ((u >> 16) & 1u);          // round-to-nearest-even
    return (unsigned short)(u >> 16);
}

__device__ __forceinline__ v8f v8f_zero() {
    v8f v;
#pragma unroll
    for (int i = 0; i < 8; ++i) v[i] = 0.f;
    return v;
}

// A-operand fragment (16x32, M x K) from LDS bf16, row-major with `stride`
// halfwords per row.  Split-K: lane<16 -> K {0..15},
// lane>=16 -> {16..31}.  Two ds_load_b128 per lane.
__device__ __forceinline__ v16bf ldsA_frag(const unsigned short* base, int stride, int lane) {
    const int r = lane & 15, hi = lane >> 4;
    const unsigned short* p = base + r * stride + hi * 16;
    BFrag f;
    f.q[0] = *reinterpret_cast<const uint4*>(p);
    f.q[1] = *reinterpret_cast<const uint4*>(p + 8);
    return f.v;
}

// B-operand fragment (32x16, K x N) sourced from a row-major [N,K] bf16
// weight array (i.e. W^T with K contiguous).  Split-K lane convention:
// lane<16 holds K = k0..k0+15 of column n=lane, lane>=16 holds K = k0+16..k0+31.
// One 32B contiguous global read per lane.
__device__ __forceinline__ v16bf glbB_frag(const unsigned short* w, int ldw,
                                           int nbase, int k0, int lane, int nlimit) {
    const int r = lane & 15, hi = lane >> 4;
    const int n = nbase + r;
    BFrag f;
    if (n < nlimit) {
        const unsigned short* p = w + (long)n * ldw + k0 + hi * 16;
        f.q[0] = *reinterpret_cast<const uint4*>(p);
        f.q[1] = *reinterpret_cast<const uint4*>(p + 8);
    } else {
        f.q[0] = make_uint4(0u, 0u, 0u, 0u);
        f.q[1] = make_uint4(0u, 0u, 0u, 0u);
    }
    return f.v;
}

__device__ __forceinline__ v8f wmma_bf16(v16bf a, v16bf b, v8f c) {
    // (neg_a, A, neg_b, B, c_mod, C, reuse_a, reuse_b)
    return __builtin_amdgcn_wmma_f32_16x16x32_bf16(false, a, false, b, (short)0, c, false, false);
}

// ---------------------------------------------------------------------------
// f32 -> bf16 conversion (weights, once per call; L2-resident afterwards)
// ---------------------------------------------------------------------------
__global__ void cvt_bf16_kernel(const float* __restrict__ src,
                                unsigned short* __restrict__ dst, int n) {
    int i = blockIdx.x * 256 + threadIdx.x;
    if (i < n) dst[i] = f2bf(src[i]);
}

// ---------------------------------------------------------------------------
// Generic WMMA GEMM:  C[M,N] = A[M,K] @ W[N,K]^T   (A,W bf16, C f32)
// Block = 256 thr (8 waves).  BM=64 A-tile double-buffered in LDS via
// async-to-LDS copies; BN=128 (one 16-wide N tile per wave, W frags
// software-pipelined from global/L2).  M % 64 == 0, K % 32 == 0.
// ---------------------------------------------------------------------------
__global__ __launch_bounds__(256) void gemm_bf16_kernel(
    const unsigned short* __restrict__ A, int lda,
    const unsigned short* __restrict__ W, int ldw,
    float* __restrict__ C, int ldc, int N, int K) {
    __shared__ __align__(16) unsigned short sA[2][64 * 40];   // 2 x (64x32 bf16, pad->40)
    const int tid  = threadIdx.x;
    const int lane = tid & 31;
    const int wave = tid >> 5;
    const int m0 = blockIdx.x * 64;
    const int nt = blockIdx.y * 128 + wave * 16;
    const int r  = tid >> 2;                 // staging row 0..63
    const int cc = (tid & 3) * 8;            // staging 16B chunk
    const unsigned short* aRow = A + (long)(m0 + r) * lda + cc;

    v8f acc[4] = {v8f_zero(), v8f_zero(), v8f_zero(), v8f_zero()};

    auto stage = [&](int buf, int k0) {
#if GEMM_ASYNC
        // LDS-direct async copy, tracked by ASYNCcnt (no VGPR round-trip)
        __builtin_amdgcn_global_load_async_to_lds_b128(
            (AS1 v4i*)(aRow + k0),
            (AS3 v4i*)&sA[buf][r * 40 + cc], 0, 0);
#else
        *reinterpret_cast<uint4*>(&sA[buf][r * 40 + cc]) =
            *reinterpret_cast<const uint4*>(aRow + k0);
#endif
    };

    stage(0, 0);
    BFrag bz; bz.q[0] = make_uint4(0u,0u,0u,0u); bz.q[1] = make_uint4(0u,0u,0u,0u);
    v16bf bfrag = bz.v;
    if (nt < N) bfrag = glbB_frag(W, ldw, nt, 0, lane, N);

    for (int k0 = 0; k0 < K; k0 += 32) {
        const int  cur     = (k0 >> 5) & 1;
        const bool hasNext = (k0 + 32) < K;
        if (hasNext) {
            stage(cur ^ 1, k0 + 32);         // prefetch next A tile
#if GEMM_ASYNC
            __builtin_amdgcn_s_wait_asynccnt(1);   // current tile landed in LDS
#endif
        } else {
#if GEMM_ASYNC
            __builtin_amdgcn_s_wait_asynccnt(0);
#endif
        }
        __syncthreads();
        if (nt < N) {
            v16bf bnext = bfrag;
            if (hasNext) bnext = glbB_frag(W, ldw, nt, k0 + 32, lane, N);  // pipeline W
#pragma unroll
            for (int mf = 0; mf < 4; ++mf) {
                const v16bf a = ldsA_frag(const_cast<const unsigned short*>(&sA[cur][mf * 16 * 40]), 40, lane);
                acc[mf] = wmma_bf16(a, bfrag, acc[mf]);
            }
            bfrag = bnext;
        }
        __syncthreads();                     // reads done before buffer reuse
    }
    if (nt < N) {
        const int rr = lane & 15, hi = lane >> 4;
        const int col = nt + rr;
        if (col < N) {
#pragma unroll
            for (int mf = 0; mf < 4; ++mf) {
                float* cp = C + (long)(m0 + mf * 16 + hi * 8) * ldc + col;
#pragma unroll
                for (int e = 0; e < 8; ++e) cp[(long)e * ldc] = acc[mf][e];
            }
        }
    }
}

// ---------------------------------------------------------------------------
// Fused PointNet encoder: one workgroup per group (4096 groups, 32 points).
// All activations live in LDS; three WMMA GEMM stages + two max-pools.
// ---------------------------------------------------------------------------
__global__ __launch_bounds__(256) void encoder_kernel(
    const float* __restrict__ pg,                                   // [4096,32,3]
    const float* __restrict__ w1, const float* __restrict__ b1,
    const float* __restrict__ g1, const float* __restrict__ bb1,    // bn1
    const unsigned short* __restrict__ w2bf, const float* __restrict__ b2,
    const unsigned short* __restrict__ w3bf, const float* __restrict__ b3,
    const float* __restrict__ g2, const float* __restrict__ bb2,    // bn2
    const unsigned short* __restrict__ w4bf, const float* __restrict__ b4,
    float* __restrict__ tokens)                                     // [4096,384]
{
    __shared__ __align__(16) unsigned short sX512[32 * 520];  // x512 bf16 (pad 512->520)
    __shared__ __align__(16) unsigned short sT[32 * 520];     // t bf16
    __shared__ __align__(16) float sF[32 * 392];              // f32 staging (f2 / f4)
    // f1 bf16 (32x128, pad->136) overlaps sF's tail; both never live together
    unsigned short* sX1 = reinterpret_cast<unsigned short*>(&sF[32 * 392]) - 32 * 136;

    const int tid = threadIdx.x, lane = tid & 31, wave = tid >> 5;
    const int g = blockIdx.x;
    const float bnS = rsqrtf(1.f + 1e-5f);

    // ---- stage 1: f1 = relu(bn1(pg @ w1^T + b1))  [32,128] -> bf16 LDS ----
    {
        const int p  = tid >> 3;          // point 0..31
        const int c0 = (tid & 7) * 16;    // 16 channels per thread
        const float x = pg[(g * 32 + p) * 3 + 0];
        const float y = pg[(g * 32 + p) * 3 + 1];
        const float z = pg[(g * 32 + p) * 3 + 2];
#pragma unroll
        for (int c = c0; c < c0 + 16; ++c) {
            float v = x * w1[c * 3 + 0] + y * w1[c * 3 + 1] + z * w1[c * 3 + 2] + b1[c];
            v = fmaxf(v * (g1[c] * bnS) + bb1[c], 0.f);
            sX1[p * 136 + c] = f2bf(v);
        }
    }
    __syncthreads();

    // ---- GEMM1: f2 = f1 @ w2^T + b2  [32,256] f32 -> sF (stride 264) ----
    {
        v8f acc[2][2] = {{v8f_zero(), v8f_zero()}, {v8f_zero(), v8f_zero()}};
        for (int k0 = 0; k0 < 128; k0 += 32) {
            const v16bf a0 = ldsA_frag(sX1 + k0, 136, lane);
            const v16bf a1 = ldsA_frag(sX1 + 16 * 136 + k0, 136, lane);
#pragma unroll
            for (int t = 0; t < 2; ++t) {
                const v16bf b = glbB_frag(w2bf, 128, (wave * 2 + t) * 16, k0, lane, 256);
                acc[0][t] = wmma_bf16(a0, b, acc[0][t]);
                acc[1][t] = wmma_bf16(a1, b, acc[1][t]);
            }
        }
        __syncthreads();   // sX1 dead, sF (overlapping) becomes live
        const int r = lane & 15, hi = lane >> 4;
#pragma unroll
        for (int t = 0; t < 2; ++t) {
            const int col = (wave * 2 + t) * 16 + r;
            const float bias = b2[col];
#pragma unroll
            for (int mf = 0; mf < 2; ++mf)
#pragma unroll
                for (int e = 0; e < 8; ++e)
                    sF[(mf * 16 + hi * 8 + e) * 264 + col] = acc[mf][t][e] + bias;
        }
    }
    __syncthreads();

    // ---- maxpool over N=32 + concat -> x512 bf16 [32,512] ----
    {
        const int c = tid;                         // 0..255
        float m = -1e30f;
        for (int r = 0; r < 32; ++r) m = fmaxf(m, sF[r * 264 + c]);
        const unsigned short mb = f2bf(m);
        for (int r = 0; r < 32; ++r) {
            sX512[r * 520 + c]        = mb;                 // broadcast global feat
            sX512[r * 520 + 256 + c]  = f2bf(sF[r * 264 + c]);
        }
    }
    __syncthreads();

    // ---- GEMM2: t = relu(bn2(x512 @ w3^T + b3))  [32,512] -> bf16 sT ----
    {
        v8f acc[2][4];
#pragma unroll
        for (int mf = 0; mf < 2; ++mf)
#pragma unroll
            for (int t = 0; t < 4; ++t) acc[mf][t] = v8f_zero();
        for (int k0 = 0; k0 < 512; k0 += 32) {
            const v16bf a0 = ldsA_frag(sX512 + k0, 520, lane);
            const v16bf a1 = ldsA_frag(sX512 + 16 * 520 + k0, 520, lane);
#pragma unroll
            for (int t = 0; t < 4; ++t) {
                const v16bf b = glbB_frag(w3bf, 512, (wave * 4 + t) * 16, k0, lane, 512);
                acc[0][t] = wmma_bf16(a0, b, acc[0][t]);
                acc[1][t] = wmma_bf16(a1, b, acc[1][t]);
            }
        }
        const int r = lane & 15, hi = lane >> 4;
#pragma unroll
        for (int t = 0; t < 4; ++t) {
            const int col = (wave * 4 + t) * 16 + r;
            const float bias = b3[col], sc = g2[col] * bnS, sh = bb2[col];
#pragma unroll
            for (int mf = 0; mf < 2; ++mf)
#pragma unroll
                for (int e = 0; e < 8; ++e) {
                    float v = acc[mf][t][e] + bias;
                    v = fmaxf(v * sc + sh, 0.f);
                    sT[(mf * 16 + hi * 8 + e) * 520 + col] = f2bf(v);
                }
        }
    }
    __syncthreads();

    // ---- GEMM3: f4 = t @ w4^T + b4  [32,384] f32 -> sF (stride 392) ----
    {
        v8f acc[2][3];
#pragma unroll
        for (int mf = 0; mf < 2; ++mf)
#pragma unroll
            for (int t = 0; t < 3; ++t) acc[mf][t] = v8f_zero();
        for (int k0 = 0; k0 < 512; k0 += 32) {
            const v16bf a0 = ldsA_frag(sT + k0, 520, lane);
            const v16bf a1 = ldsA_frag(sT + 16 * 520 + k0, 520, lane);
#pragma unroll
            for (int t = 0; t < 3; ++t) {
                const v16bf b = glbB_frag(w4bf, 512, (wave * 3 + t) * 16, k0, lane, 384);
                acc[0][t] = wmma_bf16(a0, b, acc[0][t]);
                acc[1][t] = wmma_bf16(a1, b, acc[1][t]);
            }
        }
        const int r = lane & 15, hi = lane >> 4;
#pragma unroll
        for (int t = 0; t < 3; ++t) {
            const int col = (wave * 3 + t) * 16 + r;
            const float bias = b4[col];
#pragma unroll
            for (int mf = 0; mf < 2; ++mf)
#pragma unroll
                for (int e = 0; e < 8; ++e)
                    sF[(mf * 16 + hi * 8 + e) * 392 + col] = acc[mf][t][e] + bias;
        }
    }
    __syncthreads();

    // ---- tokens = max over the 32 points ----
    for (int c = tid; c < 384; c += 256) {
        float m = -1e30f;
        for (int r = 0; r < 32; ++r) m = fmaxf(m, sF[r * 392 + c]);
        tokens[(long)g * 384 + c] = m;
    }
}

// ---------------------------------------------------------------------------
// Positional MLP on centers; accumulates into h (tokens already there).
// ---------------------------------------------------------------------------
__global__ __launch_bounds__(128) void posmlp_kernel(
    const float* __restrict__ center,
    const float* __restrict__ w1, const float* __restrict__ b1,
    const float* __restrict__ w2, const float* __restrict__ b2,
    float* __restrict__ h) {
    __shared__ float hh[128];
    const int row = blockIdx.x, t = threadIdx.x;
    const float x = center[row * 3 + 0], y = center[row * 3 + 1], z = center[row * 3 + 2];
    float v = x * w1[t * 3 + 0] + y * w1[t * 3 + 1] + z * w1[t * 3 + 2] + b1[t];
    v = 0.5f * v * (1.f + erff(v * 0.70710678118f));   // exact GELU
    hh[t] = v;
    __syncthreads();
    for (int o = t; o < 384; o += 128) {
        float s = b2[o];
        const float* w = &w2[o * 128];
        for (int k = 0; k < 128; ++k) s += hh[k] * w[k];
        h[(long)row * 384 + o] += s;
    }
}

// ---------------------------------------------------------------------------
// residual = first ? h : h + residual; hn = LN(residual) -> bf16
// ---------------------------------------------------------------------------
__global__ __launch_bounds__(384) void prenorm_kernel(
    const float* __restrict__ h, float* __restrict__ res,
    unsigned short* __restrict__ hn,
    const float* __restrict__ lnw, const float* __restrict__ lnb, int first) {
    __shared__ float sv[384];
    __shared__ float stat[2];
    const int row = blockIdx.x, t = threadIdx.x;
    float v = h[(long)row * 384 + t];
    if (!first) v += res[(long)row * 384 + t];
    res[(long)row * 384 + t] = v;
    sv[t] = v;
    __syncthreads();
    if (t == 0) {
        float s = 0.f;
        for (int i = 0; i < 384; ++i) s += sv[i];
        const float mu = s * (1.f / 384.f);
        float q = 0.f;
        for (int i = 0; i < 384; ++i) { const float d = sv[i] - mu; q += d * d; }
        stat[0] = mu;
        stat[1] = rsqrtf(q * (1.f / 384.f) + 1e-5f);
    }
    __syncthreads();
    hn[(long)row * 384 + t] = f2bf((v - stat[0]) * stat[1] * lnw[t] + lnb[t]);
}

// ---------------------------------------------------------------------------
// Causal depthwise conv1d (K=4) over L=64 + bias + SiLU; writes f32 + bf16
// ---------------------------------------------------------------------------
__global__ __launch_bounds__(256) void conv_silu_kernel(
    const float* __restrict__ xz, const float* __restrict__ cw,
    const float* __restrict__ cb, float* __restrict__ xc,
    unsigned short* __restrict__ xbf, int layer) {
    const int b = blockIdx.x / 3;
    const int d = (blockIdx.x % 3) * 256 + threadIdx.x;
    const float* w = &cw[(layer * 768 + d) * 4];
    const float w0 = w[0], w1 = w[1], w2 = w[2], w3 = w[3];
    const float bias = cb[layer * 768 + d];
    float h0 = 0.f, h1 = 0.f, h2 = 0.f;
    for (int l = 0; l < 64; ++l) {
        const float u = xz[(long)(b * 64 + l) * 1536 + d];
        float a = h0 * w0 + h1 * w1 + h2 * w2 + u * w3 + bias;
        h0 = h1; h1 = h2; h2 = u;
        a = a / (1.f + __expf(-a));                       // SiLU
        const long idx = (long)(b * 64 + l) * 768 + d;
        xc[idx]  = a;
        xbf[idx] = f2bf(a);
    }
}

// ---------------------------------------------------------------------------
// dt = softplus(dbc[:, :24] @ dt_w^T + dt_b)   [4096,768]
// ---------------------------------------------------------------------------
__global__ __launch_bounds__(256) void dtproj_kernel(
    const float* __restrict__ dbc, const float* __restrict__ dtw,
    const float* __restrict__ dtb, float* __restrict__ dt, int layer) {
    __shared__ float sr[24];
    const int row = blockIdx.x / 3;
    const int d   = (blockIdx.x % 3) * 256 + threadIdx.x;
    if (threadIdx.x < 24) sr[threadIdx.x] = dbc[(long)row * 56 + threadIdx.x];
    __syncthreads();
    const float* w = &dtw[(layer * 768 + d) * 24];
    float s = dtb[layer * 768 + d];
#pragma unroll
    for (int k = 0; k < 24; ++k) s += sr[k] * w[k];
    dt[(long)row * 768 + d] = (s > 20.f) ? s : log1pf(__expf(s));
}

// ---------------------------------------------------------------------------
// Selective scan over L=64 with fused z-gate; emits y*silu(z) as bf16.
// One thread per (b, d); B/C staged in LDS per batch row.
// ---------------------------------------------------------------------------
__global__ __launch_bounds__(256) void scan_kernel(
    const float* __restrict__ dbc, const float* __restrict__ dt,
    const float* __restrict__ xc, const float* __restrict__ xz,
    const float* __restrict__ Alog, const float* __restrict__ Dp,
    unsigned short* __restrict__ ybf, int layer) {
    __shared__ float sB[64 * 16];
    __shared__ float sC[64 * 16];
    const int b = blockIdx.x / 3;
    const int d = (blockIdx.x % 3) * 256 + threadIdx.x;
    for (int i = threadIdx.x; i < 64 * 16; i += 256) {
        const int l = i >> 4, n = i & 15;
        sB[i] = dbc[(long)(b * 64 + l) * 56 + 24 + n];
        sC[i] = dbc[(long)(b * 64 + l) * 56 + 40 + n];
    }
    __syncthreads();
    float A[16], s[16];
#pragma unroll
    for (int n = 0; n < 16; ++n) {
        A[n] = -__expf(Alog[((long)layer * 768 + d) * 16 + n]);
        s[n] = 0.f;
    }
    const float Dv = Dp[layer * 768 + d];
    for (int l = 0; l < 64; ++l) {
        const long idx = (long)(b * 64 + l) * 768 + d;
        const float dtv = dt[idx];
        const float u   = xc[idx];
        const float zv  = xz[(long)(b * 64 + l) * 1536 + 768 + d];
        float y = 0.f;
#pragma unroll
        for (int n = 0; n < 16; ++n) {
            const float dA = __expf(dtv * A[n]);
            s[n] = s[n] * dA + dtv * sB[l * 16 + n] * u;
            y += s[n] * sC[l * 16 + n];
        }
        y += u * Dv;
        y *= zv / (1.f + __expf(-zv));                    // * silu(z)
        ybf[idx] = f2bf(y);
    }
}

// ---------------------------------------------------------------------------
// residual = h + residual; out = LN(LN(residual, normf), final)
// ---------------------------------------------------------------------------
__global__ __launch_bounds__(384) void final_kernel(
    const float* __restrict__ h, const float* __restrict__ res,
    const float* __restrict__ nfw, const float* __restrict__ nfb,
    const float* __restrict__ flw, const float* __restrict__ flb,
    float* __restrict__ out) {
    __shared__ float sv[384];
    __shared__ float stat[2];
    const int row = blockIdx.x, t = threadIdx.x;
    float v = h[(long)row * 384 + t] + res[(long)row * 384 + t];
    sv[t] = v;
    __syncthreads();
    if (t == 0) {
        float s = 0.f; for (int i = 0; i < 384; ++i) s += sv[i];
        const float mu = s * (1.f / 384.f);
        float q = 0.f; for (int i = 0; i < 384; ++i) { float d = sv[i] - mu; q += d * d; }
        stat[0] = mu; stat[1] = rsqrtf(q * (1.f / 384.f) + 1e-5f);
    }
    __syncthreads();
    const float o = (v - stat[0]) * stat[1] * nfw[t] + nfb[t];
    __syncthreads();
    sv[t] = o;
    __syncthreads();
    if (t == 0) {
        float s = 0.f; for (int i = 0; i < 384; ++i) s += sv[i];
        const float mu = s * (1.f / 384.f);
        float q = 0.f; for (int i = 0; i < 384; ++i) { float d = sv[i] - mu; q += d * d; }
        stat[0] = mu; stat[1] = rsqrtf(q * (1.f / 384.f) + 1e-5f);
    }
    __syncthreads();
    out[(long)row * 384 + t] = (o - stat[0]) * stat[1] * flw[t] + flb[t];
}

// ---------------------------------------------------------------------------
// host side
// ---------------------------------------------------------------------------
extern "C" void kernel_launch(void* const* d_in, const int* in_sizes, int n_in,
                              void* d_out, int out_size, void* d_ws, size_t ws_size,
                              hipStream_t stream) {
    // inputs, flattened leaf-order of setup_inputs()
    const float* pg      = (const float*)d_in[0];    // [64,64,32,3]
    const float* center  = (const float*)d_in[1];    // [64,64,3]
    const float* e_w1    = (const float*)d_in[2];
    const float* e_b1    = (const float*)d_in[3];
    const float* e_bn1g  = (const float*)d_in[4];
    const float* e_bn1b  = (const float*)d_in[5];
    const float* e_w2    = (const float*)d_in[6];
    const float* e_b2    = (const float*)d_in[7];
    const float* e_w3    = (const float*)d_in[8];
    const float* e_b3    = (const float*)d_in[9];
    const float* e_bn2g  = (const float*)d_in[10];
    const float* e_bn2b  = (const float*)d_in[11];
    const float* e_w4    = (const float*)d_in[12];
    const float* e_b4    = (const float*)d_in[13];
    const float* p_w1    = (const float*)d_in[14];
    const float* p_b1    = (const float*)d_in[15];
    const float* p_w2    = (const float*)d_in[16];
    const float* p_b2    = (const float*)d_in[17];
    const float* ln_w    = (const float*)d_in[18];   // [12,384]
    const float* ln_b    = (const float*)d_in[19];
    const float* in_w    = (const float*)d_in[20];   // [12,1536,384]
    const float* conv_w  = (const float*)d_in[21];   // [12,768,4]
    const float* conv_b  = (const float*)d_in[22];
    const float* xproj_w = (const float*)d_in[23];   // [12,56,768]
    const float* dt_w    = (const float*)d_in[24];   // [12,768,24]
    const float* dt_b    = (const float*)d_in[25];
    const float* A_log   = (const float*)d_in[26];   // [12,768,16]
    const float* Dp      = (const float*)d_in[27];
    const float* out_w   = (const float*)d_in[28];   // [12,384,768]
    const float* normf_w = (const float*)d_in[29];
    const float* normf_b = (const float*)d_in[30];
    const float* final_w = (const float*)d_in[31];
    const float* final_b = (const float*)d_in[32];

    // workspace carving (~103 MB)
    char* wsp = (char*)d_ws;
    auto carve = [&](size_t bytes) -> char* {
        char* p = wsp;
        wsp += (bytes + 255) & ~(size_t)255;
        return p;
    };
    unsigned short* w2bf   = (unsigned short*)carve(256 * 128 * 2);
    unsigned short* w3bf   = (unsigned short*)carve(512 * 512 * 2);
    unsigned short* w4bf   = (unsigned short*)carve(384 * 512 * 2);
    unsigned short* inwbf  = (unsigned short*)carve((size_t)12 * 1536 * 384 * 2);
    unsigned short* xpwbf  = (unsigned short*)carve((size_t)12 * 56 * 768 * 2);
    unsigned short* outwbf = (unsigned short*)carve((size_t)12 * 384 * 768 * 2);
    float*          bufH   = (float*)carve((size_t)4096 * 384 * 4);
    float*          bufRes = (float*)carve((size_t)4096 * 384 * 4);
    unsigned short* bufHn  = (unsigned short*)carve((size_t)4096 * 384 * 2);
    float*          bufXZ  = (float*)carve((size_t)4096 * 1536 * 4);
    float*          bufXc  = (float*)carve((size_t)4096 * 768 * 4);
    unsigned short* bufXbf = (unsigned short*)carve((size_t)4096 * 768 * 2);
    float*          bufDbc = (float*)carve((size_t)4096 * 56 * 4);
    float*          bufDt  = (float*)carve((size_t)4096 * 768 * 4);
    unsigned short* bufYbf = (unsigned short*)carve((size_t)4096 * 768 * 2);

    auto cvt = [&](const float* s, unsigned short* d, int n) {
        cvt_bf16_kernel<<<(n + 255) / 256, 256, 0, stream>>>(s, d, n);
    };
    cvt(e_w2, w2bf, 256 * 128);
    cvt(e_w3, w3bf, 512 * 512);
    cvt(e_w4, w4bf, 384 * 512);
    cvt(in_w, inwbf, 12 * 1536 * 384);
    cvt(xproj_w, xpwbf, 12 * 56 * 768);
    cvt(out_w, outwbf, 12 * 384 * 768);

    // encoder -> tokens in bufH, then add positional MLP
    encoder_kernel<<<4096, 256, 0, stream>>>(pg, e_w1, e_b1, e_bn1g, e_bn1b,
                                             w2bf, e_b2, w3bf, e_b3, e_bn2g, e_bn2b,
                                             w4bf, e_b4, bufH);
    posmlp_kernel<<<4096, 128, 0, stream>>>(center, p_w1, p_b1, p_w2, p_b2, bufH);

    for (int i = 0; i < 12; ++i) {
        prenorm_kernel<<<4096, 384, 0, stream>>>(bufH, bufRes, bufHn,
                                                 ln_w + i * 384, ln_b + i * 384, i == 0);
        // xz = hn @ in_w^T : [4096,1536], K=384
        gemm_bf16_kernel<<<dim3(64, 12), 256, 0, stream>>>(
            bufHn, 384, inwbf + (size_t)i * 1536 * 384, 384, bufXZ, 1536, 1536, 384);
        conv_silu_kernel<<<64 * 3, 256, 0, stream>>>(bufXZ, conv_w, conv_b, bufXc, bufXbf, i);
        // dbc = x @ xproj_w^T : [4096,56], K=768
        gemm_bf16_kernel<<<dim3(64, 1), 256, 0, stream>>>(
            bufXbf, 768, xpwbf + (size_t)i * 56 * 768, 768, bufDbc, 56, 56, 768);
        dtproj_kernel<<<4096 * 3, 256, 0, stream>>>(bufDbc, dt_w, dt_b, bufDt, i);
        scan_kernel<<<64 * 3, 256, 0, stream>>>(bufDbc, bufDt, bufXc, bufXZ, A_log, Dp, bufYbf, i);
        // h = y @ out_w^T : [4096,384], K=768
        gemm_bf16_kernel<<<dim3(64, 3), 256, 0, stream>>>(
            bufYbf, 768, outwbf + (size_t)i * 384 * 768, 768, bufH, 384, 384, 768);
    }

    final_kernel<<<4096, 384, 0, stream>>>(bufH, bufRes, normf_w, normf_b,
                                           final_w, final_b, (float*)d_out);
}
